// mLSTMCell_86603720556996
// MI455X (gfx1250) — matrix operand order repaired and checked by reading, not compiled, not run on hardware
//
#include <hip/hip_runtime.h>
#include <hip/hip_bf16.h>
#include <math.h>

// ---------------- problem constants (from reference setup_inputs) ----------
#define BB   2
#define SS   2048
#define EE   1024
#define NHH  4
#define DHH  256     // EE / NHH
#define E3   3072    // 3*EE

#define NEG_INF (-__builtin_inff())

typedef __attribute__((ext_vector_type(16))) __bf16 v16bf;
typedef __attribute__((ext_vector_type(8)))  float  v8f;

union F16x16 { v16bf v; unsigned u[8]; };

// Pack two f32 into packed bf16 (round-half-up): 1 add per float + 1 v_perm.
__device__ __forceinline__ unsigned pk_bf16(float a, float b) {
    unsigned xa = __float_as_uint(a) + 0x8000u;
    unsigned xb = __float_as_uint(b) + 0x8000u;
#if __has_builtin(__builtin_amdgcn_perm)
    // result = { hi16(xb), hi16(xa) }
    return __builtin_amdgcn_perm(xb, xa, 0x07060302u);
#else
    return (xa >> 16) | (xb & 0xFFFF0000u);
#endif
}

// reductions over the 16 lanes of a lane-half (xor masks < 16 stay in-half)
__device__ __forceinline__ float hred_max(float x) {
    x = fmaxf(x, __shfl_xor(x, 1));
    x = fmaxf(x, __shfl_xor(x, 2));
    x = fmaxf(x, __shfl_xor(x, 4));
    x = fmaxf(x, __shfl_xor(x, 8));
    return x;
}
__device__ __forceinline__ float hred_sum(float x) {
    x += __shfl_xor(x, 1);
    x += __shfl_xor(x, 2);
    x += __shfl_xor(x, 4);
    x += __shfl_xor(x, 8);
    return x;
}

// ---------------------------------------------------------------------------
// Kernel 1: gate pre-activations.  One 128-thread block per (b,s) row.
// ig_out/fg_out layout: [(b*NH+h)*S + s]
// ---------------------------------------------------------------------------
__global__ __launch_bounds__(128) void gates_kernel(
    const float* __restrict__ q, const float* __restrict__ k,
    const float* __restrict__ v,
    const float* __restrict__ igk, const float* __restrict__ igb,
    const float* __restrict__ fgk, const float* __restrict__ fgb,
    float* __restrict__ ig_out, float* __restrict__ fg_out)
{
    const int bs = blockIdx.x;          // = b*SS + s
    const int b  = bs / SS;
    const int s  = bs % SS;
    const float* qr = q + (size_t)bs * EE;
    const float* kr = k + (size_t)bs * EE;
    const float* vr = v + (size_t)bs * EE;

    float aig[4] = {0.f, 0.f, 0.f, 0.f};
    float afg[4] = {0.f, 0.f, 0.f, 0.f};

    for (int e = threadIdx.x; e < E3; e += 128) {
        float x = (e < EE) ? qr[e] : ((e < 2 * EE) ? kr[e - EE] : vr[e - 2 * EE]);
        const float4 wi = *(const float4*)(igk + (size_t)e * 4);
        const float4 wf = *(const float4*)(fgk + (size_t)e * 4);
        aig[0] += x * wi.x; aig[1] += x * wi.y; aig[2] += x * wi.z; aig[3] += x * wi.w;
        afg[0] += x * wf.x; afg[1] += x * wf.y; afg[2] += x * wf.z; afg[3] += x * wf.w;
    }
    #pragma unroll
    for (int h = 0; h < 4; ++h) {
        #pragma unroll
        for (int m = 1; m < 32; m <<= 1) {
            aig[h] += __shfl_xor(aig[h], m);
            afg[h] += __shfl_xor(afg[h], m);
        }
    }
    __shared__ float red[4][8];
    const int wave = threadIdx.x >> 5;
    const int lane = threadIdx.x & 31;
    if (lane == 0) {
        #pragma unroll
        for (int h = 0; h < 4; ++h) { red[wave][h] = aig[h]; red[wave][4 + h] = afg[h]; }
    }
    __syncthreads();
    if (threadIdx.x < 8) {
        float t = red[0][threadIdx.x] + red[1][threadIdx.x] +
                  red[2][threadIdx.x] + red[3][threadIdx.x];
        int h = threadIdx.x & 3;
        size_t idx = ((size_t)(b * NHH + h)) * SS + s;
        if (threadIdx.x < 4) ig_out[idx] = t + igb[h];
        else                 fg_out[idx] = t + fgb[h];
    }
}

// ---------------------------------------------------------------------------
// Kernel 2: inclusive cumsum of log_sigmoid(fg) per (b,h).  8 serial threads.
// ---------------------------------------------------------------------------
__global__ void scan_kernel(const float* __restrict__ fg_raw, float* __restrict__ Acum)
{
    int bh = threadIdx.x;
    if (bh >= BB * NHH) return;
    float c = 0.f;
    for (int s = 0; s < SS; ++s) {
        float x = fg_raw[(size_t)bh * SS + s];
        // stable log_sigmoid
        float ls = fminf(x, 0.f) - log1pf(__builtin_expf(-fabsf(x)));
        c += ls;
        Acum[(size_t)bh * SS + s] = c;
    }
}

// ---------------------------------------------------------------------------
// Kernel 2b: K f32 -> bf16, same (b,s,e) layout.  8 elems / thread, coalesced.
// ---------------------------------------------------------------------------
__global__ __launch_bounds__(256) void cvt_k_kernel(const float* __restrict__ src,
                                                    unsigned short* __restrict__ dst)
{
    size_t i = ((size_t)blockIdx.x * 256 + threadIdx.x) * 8;
    const float4 a = *(const float4*)(src + i);
    const float4 b = *(const float4*)(src + i + 4);
    uint4 o;
    o.x = pk_bf16(a.x, a.y); o.y = pk_bf16(a.z, a.w);
    o.z = pk_bf16(b.x, b.y); o.w = pk_bf16(b.z, b.w);
    *(uint4*)(dst + i) = o;
}

// ---------------------------------------------------------------------------
// Kernel 2c: V f32 (b,s,h,d) -> bf16 transposed vt[(b*NH+h)*DH + d][s].
// Thread: one d, 8 consecutive s.  Reads coalesce across threads (same s).
// ---------------------------------------------------------------------------
__global__ __launch_bounds__(64) void cvt_v_kernel(const float* __restrict__ v,
                                                   unsigned short* __restrict__ vt)
{
    const int bh = blockIdx.z;
    const int b  = bh / NHH, hh = bh % NHH;
    const int d  = blockIdx.y * 64 + threadIdx.x;
    const int s0 = blockIdx.x * 8;
    const float* sp = v + ((size_t)(b * SS + s0)) * EE + hh * DHH + d;
    float x[8];
    #pragma unroll
    for (int i = 0; i < 8; ++i) x[i] = sp[(size_t)i * EE];
    uint4 o;
    o.x = pk_bf16(x[0], x[1]); o.y = pk_bf16(x[2], x[3]);
    o.z = pk_bf16(x[4], x[5]); o.w = pk_bf16(x[6], x[7]);
    *(uint4*)(vt + ((size_t)bh * DHH + d) * SS + s0) = o;
}

// ---------------------------------------------------------------------------
// Kernel 3: flash-style mLSTM attention + output layernorm.
// grid = (SS/64, BB*NHH), block = 128 (4 waves).  Wave w owns 16 query rows.
// Q A-fragments pre-packed in LDS; K/V pre-converted to bf16 (V transposed),
// so the inner loop is pure b128 loads + v_wmma.
// ---------------------------------------------------------------------------
__global__ __launch_bounds__(128) void mlstm_attn_kernel(
    const float* __restrict__ q,
    const unsigned short* __restrict__ kbf,   // bf16, (b,s,e) layout
    const unsigned short* __restrict__ vt,    // bf16, (b*NH+h, d, s) layout
    const float* __restrict__ ig, const float* __restrict__ Acum,
    const float* __restrict__ nw, float* __restrict__ out)
{
    __shared__ unsigned ldsQ[4][8][32][8];  // 32 KB: pre-packed Q A-fragments
    __shared__ float    ldsP[4][16 * 32];   //  8 KB: per-wave P-tile transpose

    const int wave = threadIdx.x >> 5;
    const int lane = threadIdx.x & 31;
    const int n    = lane & 15;             // column / row-in-fragment id
    const int hi   = lane >> 4;             // lane-half
    const int lb   = hi * 8;                // A-layout K base within 32-block
    const int bh   = blockIdx.y;
    const int b    = bh / NHH;
    const int hh   = bh % NHH;
    const int qbase = blockIdx.x * 64 + wave * 16;
    const size_t bhS = (size_t)bh * SS;

    const float LOG2E = 1.4426950408889634f;
    const float SC    = 1.0f / 16.0f;       // 1/sqrt(DH)

    // ---- pack Q fragments (A-layout 16x32 bf16) into per-wave LDS ----------
    {
        const float* qrow = q + ((size_t)(b * SS + qbase + n)) * EE + hh * DHH;
        #pragma unroll
        for (int f = 0; f < 8; ++f) {
            const float4 r0 = *(const float4*)(qrow + f * 32 + lb);
            const float4 r1 = *(const float4*)(qrow + f * 32 + lb + 4);
            const float4 r2 = *(const float4*)(qrow + f * 32 + 16 + lb);
            const float4 r3 = *(const float4*)(qrow + f * 32 + 16 + lb + 4);
            uint4 lo, hi4;
            lo.x  = pk_bf16(r0.x, r0.y); lo.y  = pk_bf16(r0.z, r0.w);
            lo.z  = pk_bf16(r1.x, r1.y); lo.w  = pk_bf16(r1.z, r1.w);
            hi4.x = pk_bf16(r2.x, r2.y); hi4.y = pk_bf16(r2.z, r2.w);
            hi4.z = pk_bf16(r3.x, r3.y); hi4.w = pk_bf16(r3.z, r3.w);
            *(uint4*)&ldsQ[wave][f][lane][0] = lo;
            *(uint4*)&ldsQ[wave][f][lane][4] = hi4;
        }
    }
    asm volatile("s_wait_dscnt 0" ::: "memory");

    float Arow[8];
    #pragma unroll
    for (int r = 0; r < 8; ++r) Arow[r] = Acum[bhS + qbase + r + 8 * hi];

    v8f zf = {};
    v8f acc[16];
    #pragma unroll
    for (int t = 0; t < 16; ++t) acc[t] = zf;

    float m_run[8], l_run[8];
    #pragma unroll
    for (int r = 0; r < 8; ++r) { m_run[r] = NEG_INF; l_run[r] = 0.f; }

    float* Pw = ldsP[wave];

    // ---- causal key loop, 32 keys per tile ---------------------------------
    for (int kb0 = 0; kb0 <= qbase + 15; kb0 += 32) {
        // QK^T: two 16x16 S-tiles, K=256 contraction in 8 bf16 WMMA steps each
        v8f s0 = zf, s1 = zf;
        const unsigned short* kb_ = kbf + ((size_t)(b * SS + kb0 + n)) * EE + hh * DHH + hi * 8;
        #pragma unroll
        for (int ks = 0; ks < 8; ++ks) {
            F16x16 qf, kf0, kf1;
            {
                const uint4* qp = (const uint4*)&ldsQ[wave][ks][lane][0];
                uint4 qlo = qp[0], qhi = qp[1];
                qf.u[0] = qlo.x; qf.u[1] = qlo.y; qf.u[2] = qlo.z; qf.u[3] = qlo.w;
                qf.u[4] = qhi.x; qf.u[5] = qhi.y; qf.u[6] = qhi.z; qf.u[7] = qhi.w;
            }
            {
                const uint4* kp0 = (const uint4*)(kb_ + ks * 32);
                const uint4* kp1 = (const uint4*)(kb_ + (size_t)16 * EE + ks * 32);
                uint4 c0 = kp0[0], c1 = kp0[2];
                uint4 d0 = kp1[0], d1 = kp1[2];
                kf0.u[0] = c0.x; kf0.u[1] = c0.y; kf0.u[2] = c0.z; kf0.u[3] = c0.w;
                kf0.u[4] = c1.x; kf0.u[5] = c1.y; kf0.u[6] = c1.z; kf0.u[7] = c1.w;
                kf1.u[0] = d0.x; kf1.u[1] = d0.y; kf1.u[2] = d0.z; kf1.u[3] = d0.w;
                kf1.u[4] = d1.x; kf1.u[5] = d1.y; kf1.u[6] = d1.z; kf1.u[7] = d1.w;
            }
            s0 = __builtin_amdgcn_wmma_f32_16x16x32_bf16(
                     false, qf.v, false, kf0.v, (short)0, s0, false, false);
            s1 = __builtin_amdgcn_wmma_f32_16x16x32_bf16(
                     false, qf.v, false, kf1.v, (short)0, s1, false, false);
        }

        // decay gating + online stabilization (C-layout: row = qbase+r+8*hi, col = kb0+jt*16+n)
        const int j0 = kb0 + n, j1 = j0 + 16;
        const float Aj0 = Acum[bhS + j0], g0 = ig[bhS + j0];
        const float Aj1 = Acum[bhS + j1], g1 = ig[bhS + j1];
        v8f scv;
        #pragma unroll
        for (int r = 0; r < 8; ++r) {
            const int irow = qbase + r + 8 * hi;
            float ld0 = (j0 <= irow) ? (Arow[r] - Aj0 + g0) : NEG_INF;
            float ld1 = (j1 <= irow) ? (Arow[r] - Aj1 + g1) : NEG_INF;
            float tm  = hred_max(fmaxf(ld0, ld1));
            float mn  = fmaxf(m_run[r], tm);
            float scale = __builtin_exp2f((m_run[r] - mn) * LOG2E);
            float p0 = s0[r] * SC * __builtin_exp2f((ld0 - mn) * LOG2E);
            float p1 = s1[r] * SC * __builtin_exp2f((ld1 - mn) * LOG2E);
            l_run[r] = l_run[r] * scale + hred_sum(p0 + p1);
            m_run[r] = mn;
            scv[r]   = scale;
            Pw[(r + 8 * hi) * 32 + n]      = p0;   // stash P tile (D-layout)
            Pw[(r + 8 * hi) * 32 + n + 16] = p1;
        }
        #pragma unroll
        for (int t = 0; t < 16; ++t) acc[t] *= scv;

        asm volatile("s_wait_dscnt 0" ::: "memory");

        // re-read P in A-layout (row m = n, K = j in [0,32))
        F16x16 pf;
        #pragma unroll
        for (int i = 0; i < 8; ++i) {
            const int e0 = 2 * i, e1 = 2 * i + 1;
            const int ja = (e0 < 8) ? (lb + e0) : (16 + lb + (e0 - 8));
            const int jb = (e1 < 8) ? (lb + e1) : (16 + lb + (e1 - 8));
            pf.u[i] = pk_bf16(Pw[n * 32 + ja], Pw[n * 32 + jb]);
        }

        // P·V: 16 output N-tiles, full K=32 contraction per WMMA (bf16 V^T)
        const unsigned short* vbase = vt + (size_t)bh * DHH * SS + kb0 + hi * 8;
        #pragma unroll
        for (int t = 0; t < 16; ++t) {
            F16x16 vf;
            {
                const uint4* vp = (const uint4*)(vbase + (size_t)(t * 16 + n) * SS);
                uint4 a = vp[0], c = vp[2];
                vf.u[0] = a.x; vf.u[1] = a.y; vf.u[2] = a.z; vf.u[3] = a.w;
                vf.u[4] = c.x; vf.u[5] = c.y; vf.u[6] = c.z; vf.u[7] = c.w;
            }
            acc[t] = __builtin_amdgcn_wmma_f32_16x16x32_bf16(
                         false, pf.v, false, vf.v, (short)0, acc[t], false, false);
        }
    }

    // ---- epilogue: normalize + per-head layernorm + store ------------------
    float wv[16];
    #pragma unroll
    for (int t = 0; t < 16; ++t) wv[t] = nw[hh * DHH + t * 16 + n];

    #pragma unroll
    for (int r = 0; r < 8; ++r) {
        float norm = fmaxf(fabsf(l_run[r]), __builtin_exp2f(-m_run[r] * LOG2E));
        float invn = 1.0f / (norm + 1e-6f);
        float hv[16];
        float s1 = 0.f, s2 = 0.f;
        #pragma unroll
        for (int t = 0; t < 16; ++t) {
            hv[t] = acc[t][r] * invn;
            s1 += hv[t];
            s2 += hv[t] * hv[t];
        }
        s1 = hred_sum(s1);
        s2 = hred_sum(s2);
        float mean = s1 * (1.0f / 256.0f);
        float var  = s2 * (1.0f / 256.0f) - mean * mean;
        float rstd = rsqrtf(var + 1e-6f);
        const int si = qbase + r + 8 * hi;
        float* orow = out + ((size_t)(b * SS + si)) * EE + hh * DHH + n;
        #pragma unroll
        for (int t = 0; t < 16; ++t)
            orow[t * 16] = (hv[t] - mean) * rstd * wv[t];
    }
}

// ---------------------------------------------------------------------------
extern "C" void kernel_launch(void* const* d_in, const int* in_sizes, int n_in,
                              void* d_out, int out_size, void* d_ws, size_t ws_size,
                              hipStream_t stream)
{
    const float* q   = (const float*)d_in[0];
    const float* k   = (const float*)d_in[1];
    const float* v   = (const float*)d_in[2];
    const float* igk = (const float*)d_in[3];
    const float* igb = (const float*)d_in[4];
    const float* fgk = (const float*)d_in[5];
    const float* fgb = (const float*)d_in[6];
    const float* nw  = (const float*)d_in[7];
    float* out = (float*)d_out;

    const size_t NG = (size_t)BB * NHH * SS;        // gate array elems
    const size_t NE = (size_t)BB * SS * EE;         // full tensor elems

    float* ws     = (float*)d_ws;
    float* ig_arr = ws;                             // NG floats
    float* fg_raw = ws + NG;                        // NG floats
    float* Acum   = ws + 2 * NG;                    // NG floats
    unsigned short* kbf = (unsigned short*)(ws + 3 * NG);   // NE bf16
    unsigned short* vtr = kbf + NE;                 // NE bf16

    gates_kernel<<<dim3(BB * SS), 128, 0, stream>>>(q, k, v, igk, igb, fgk, fgb,
                                                    ig_arr, fg_raw);
    scan_kernel<<<1, 32, 0, stream>>>(fg_raw, Acum);
    cvt_k_kernel<<<dim3((unsigned)(NE / (256 * 8))), 256, 0, stream>>>(k, kbf);
    cvt_v_kernel<<<dim3(SS / 8, DHH / 64, BB * NHH), 64, 0, stream>>>(v, vtr);
    mlstm_attn_kernel<<<dim3(SS / 64, BB * NHH), 128, 0, stream>>>(
        q, kbf, vtr, ig_arr, Acum, nw, out);
}
